// TranscendentNeuralNetwork_50964081934959
// MI455X (gfx1250) — matrix operand order repaired and checked by reading, not compiled, not run on hardware
//
#include <hip/hip_runtime.h>
#include <hip/hip_bf16.h>
#include <math.h>

// ---------------------------------------------------------------------------
// Types
// ---------------------------------------------------------------------------
typedef __attribute__((ext_vector_type(16))) __bf16        v16bf;
typedef __attribute__((ext_vector_type(8)))  float         v8f;
typedef __attribute__((ext_vector_type(4)))  unsigned int  u32x4;

struct Frag32B { u32x4 lo, hi; };   // 32 bytes, bit_cast source for v16bf

__device__ __forceinline__ unsigned short f2bf(float f) {
    unsigned int u = __float_as_uint(f);
    unsigned int r = u + 0x7FFFu + ((u >> 16) & 1u);   // round-to-nearest-even
    return (unsigned short)(r >> 16);
}

// ---------------------------------------------------------------------------
// f32 -> bf16, 4 elements/thread (n multiple of 1024). Used for x.
// ---------------------------------------------------------------------------
__global__ __launch_bounds__(256)
void k_cvt(const float* __restrict__ src, unsigned short* __restrict__ dst) {
    size_t i = ((size_t)blockIdx.x * 256 + threadIdx.x) * 4;
    float4 v = *(const float4*)(src + i);
    dst[i + 0] = f2bf(v.x);
    dst[i + 1] = f2bf(v.y);
    dst[i + 2] = f2bf(v.z);
    dst[i + 3] = f2bf(v.w);
}

// ---------------------------------------------------------------------------
// f32 -> bf16 + transpose: src is K x N (row-major), dst is N x K (row-major).
// 32x32 LDS-tiled, block (32,8). One-time cost for the weight matrices so the
// GEMM's B staging becomes a pure coalesced copy (no scalar b16 scatter).
// ---------------------------------------------------------------------------
__global__ __launch_bounds__(256)
void k_cvt_t(const float* __restrict__ src, unsigned short* __restrict__ dst,
             int N, int K) {
    __shared__ unsigned short tile[32][33];
    const int n0 = blockIdx.x * 32;
    const int k0 = blockIdx.y * 32;
    const int tx = threadIdx.x;          // 0..31
    const int ty = threadIdx.y;          // 0..7
    #pragma unroll
    for (int i = 0; i < 4; ++i) {
        int k = ty + 8 * i;
        tile[k][tx] = f2bf(src[(size_t)(k0 + k) * N + n0 + tx]);
    }
    __syncthreads();
    #pragma unroll
    for (int i = 0; i < 4; ++i) {
        int n = ty + 8 * i;
        dst[(size_t)(n0 + n) * K + k0 + tx] = tile[tx][n];
    }
}

// ---------------------------------------------------------------------------
// Transcendent probs: one block per row r. Row state (1024 f32) in LDS.
// 7 depths x 3 gates, gate: new[j] = c*st[j] - s*st[j+1] + s*st[j-1] (circular)
// probs[r] = st[0]^2
// ---------------------------------------------------------------------------
__global__ __launch_bounds__(256)
void k_probs(const float* __restrict__ tw, float* __restrict__ probs) {
    __shared__ float row[1024];
    const int r = blockIdx.x;
    const int t = threadIdx.x;
    #pragma unroll
    for (int i = 0; i < 4; ++i) row[t + 256 * i] = 0.03125f;   // 1/sqrt(1024)
    __syncthreads();

    for (int d = 0; d < 7; ++d) {
        #pragma unroll
        for (int k = 0; k < 3; ++k) {
            float ang = tw[((size_t)d * 1024 + r) * 1024 + k];
            float c = cosf(ang), s = sinf(ang);
            float nv[4];
            #pragma unroll
            for (int i = 0; i < 4; ++i) {
                int j  = t + 256 * i;
                int jp = (j + 1) & 1023;
                int jm = (j + 1023) & 1023;
                nv[i] = c * row[j] - s * row[jp] + s * row[jm];
            }
            __syncthreads();
            #pragma unroll
            for (int i = 0; i < 4; ++i) row[t + 256 * i] = nv[i];
            __syncthreads();
        }
    }
    if (t == 0) { float v = row[0]; probs[r] = v * v; }
}

// ---------------------------------------------------------------------------
// Tiled bf16 WMMA GEMM:  D[M,N] = epilogue(A[M,K] @ B[K,N] + bias (+extra))
// B is supplied pre-transposed (Bt: N x K), so both A and B staging are pure
// coalesced b128 copies into LDS. Block tile 128x128, 8 waves (4M x 2N),
// wave tile 32x64 = 8 WMMA accumulators. K staged 32-wide.
// MODE 0: relu, bf16 out | MODE 2: tanh(+extra[n]), bf16 out | MODE 3: relu, f32 out
// ---------------------------------------------------------------------------
#define BM 128
#define BN 128
#define BK 32
#define STA 40   // LDS stride (elems): 80 bytes = 5*16B -> aligned, bank-spread
#define STB 40

template <int MODE>
__global__ __launch_bounds__(256)
void gemm_bf16(const unsigned short* __restrict__ A,    // M x K, row-major bf16
               const unsigned short* __restrict__ Bt,   // N x K, row-major bf16
               const float* __restrict__ bias,          // N
               const float* __restrict__ extra,         // N (probs) or nullptr
               void* __restrict__ D,                    // M x N
               int M, int N, int K) {
    __shared__ unsigned short As[BM * STA];
    __shared__ unsigned short Bs[BN * STB];

    const int tid  = threadIdx.x;
    const int wave = tid >> 5;
    const int lane = tid & 31;
    const int lm   = lane & 15;
    const int hi   = lane >> 4;          // half-wave select
    const int wm   = wave >> 1;          // 0..3  (32 rows each)
    const int wn   = wave & 1;           // 0..1  (64 cols each)

    const int m0 = blockIdx.x * BM;
    const int n0 = blockIdx.y * BN;

    const v8f vz = {0.f, 0.f, 0.f, 0.f, 0.f, 0.f, 0.f, 0.f};
    v8f acc[2][4];
    #pragma unroll
    for (int i = 0; i < 2; ++i)
        #pragma unroll
        for (int j = 0; j < 4; ++j) acc[i][j] = vz;

    // staging roles (fixed per thread): 128 rows x 32 k, 16 elems/thread
    const int srow = tid >> 1;            // 0..127
    const int skb  = (tid & 1) * 16;      // 0 or 16

    for (int k0 = 0; k0 < K; k0 += BK) {
        // ---- global loads (overlap with previous tile's compute) ----
        const unsigned short* ag = A  + (size_t)(m0 + srow) * K + k0 + skb;
        u32x4 a0 = *(const u32x4*)(ag);
        u32x4 a1 = *(const u32x4*)(ag + 8);

        const unsigned short* bg = Bt + (size_t)(n0 + srow) * K + k0 + skb;
        u32x4 b0 = *(const u32x4*)(bg);
        u32x4 b1 = *(const u32x4*)(bg + 8);

        __syncthreads();   // previous compute done reading LDS

        *(u32x4*)(As + srow * STA + skb)     = a0;
        *(u32x4*)(As + srow * STA + skb + 8) = a1;
        *(u32x4*)(Bs + srow * STB + skb)     = b0;
        *(u32x4*)(Bs + srow * STB + skb + 8) = b1;

        __syncthreads();

        // ---- fragments per CDNA5 §7.12.2 VGPR layouts ----
        v16bf afr[2];
        #pragma unroll
        for (int fi = 0; fi < 2; ++fi) {
            int row = wm * 32 + fi * 16 + lm;
            int kbA = hi ? 8 : 0;         // lanes<16: K{0-7,16-23}; >=16: K{8-15,24-31}
            Frag32B f;
            f.lo = *(const u32x4*)(As + row * STA + kbA);
            f.hi = *(const u32x4*)(As + row * STA + kbA + 16);
            afr[fi] = __builtin_bit_cast(v16bf, f);
        }
        v16bf bfr[4];
        #pragma unroll
        for (int fj = 0; fj < 4; ++fj) {
            int col = wn * 64 + fj * 16 + lm;
            int kbB = hi ? 8 : 0;         // lanes<16: K{0-7,16-23}; >=16: K{8-15,24-31}
            Frag32B f;
            f.lo = *(const u32x4*)(Bs + col * STB + kbB);
            f.hi = *(const u32x4*)(Bs + col * STB + kbB + 16);
            bfr[fj] = __builtin_bit_cast(v16bf, f);
        }

        #pragma unroll
        for (int fi = 0; fi < 2; ++fi)
            #pragma unroll
            for (int fj = 0; fj < 4; ++fj)
                acc[fi][fj] = __builtin_amdgcn_wmma_f32_16x16x32_bf16(
                    false, afr[fi], false, bfr[fj],
                    (short)0, acc[fi][fj], false, false);
    }

    // ---- epilogue ----
    #pragma unroll
    for (int fi = 0; fi < 2; ++fi) {
        #pragma unroll
        for (int fj = 0; fj < 4; ++fj) {
            int n = n0 + wn * 64 + fj * 16 + lm;
            float bval = bias[n];
            float eval = (MODE == 2) ? extra[n] : 0.f;
            #pragma unroll
            for (int v = 0; v < 8; ++v) {
                int m = m0 + wm * 32 + fi * 16 + v + 8 * hi;
                float val = acc[fi][fj][v] + bval;
                if (MODE == 2) val = tanhf(val + eval);
                else           val = fmaxf(val, 0.f);
                if (MODE == 3) ((float*)D)[(size_t)m * N + n] = val;
                else ((unsigned short*)D)[(size_t)m * N + n] = f2bf(val);
            }
        }
    }
}

// ---------------------------------------------------------------------------
// Launch: probs + conversions (weights transposed), then 4 WMMA GEMM layers
// ---------------------------------------------------------------------------
extern "C" void kernel_launch(void* const* d_in, const int* in_sizes, int n_in,
                              void* d_out, int out_size, void* d_ws, size_t ws_size,
                              hipStream_t stream) {
    (void)in_sizes; (void)n_in; (void)out_size; (void)ws_size;
    const float* x    = (const float*)d_in[0];
    const float* w0   = (const float*)d_in[1];
    const float* b0   = (const float*)d_in[2];
    const float* w1   = (const float*)d_in[3];
    const float* b1   = (const float*)d_in[4];
    const float* tw   = (const float*)d_in[5];
    const float* cw   = (const float*)d_in[6];
    const float* cb   = (const float*)d_in[7];
    const float* wout = (const float*)d_in[8];
    const float* bout = (const float*)d_in[9];

    const int Bq = 16384, H = 1024, DO = 256;

    char* ws = (char*)d_ws;
    unsigned short* xb    = (unsigned short*)ws;  ws += (size_t)Bq * H * 2;  // 32MB
    unsigned short* h1b   = (unsigned short*)ws;  ws += (size_t)Bq * H * 2;  // 32MB
    unsigned short* w0t   = (unsigned short*)ws;  ws += (size_t)H * H * 2;
    unsigned short* w1t   = (unsigned short*)ws;  ws += (size_t)H * H * 2;
    unsigned short* cwt   = (unsigned short*)ws;  ws += (size_t)H * H * 2;
    unsigned short* woutt = (unsigned short*)ws;  ws += (size_t)H * DO * 2;
    float*          probs = (float*)ws;           ws += (size_t)H * 4;
    unsigned short* h2b = xb;    // reuse: x consumed by layer 0
    unsigned short* h3b = h1b;   // reuse: h1 consumed by layer 1

    // conversions: x straight; weights converted + transposed to N x K
    k_cvt<<<(Bq * H) / 1024, 256, 0, stream>>>(x, xb);
    k_cvt_t<<<dim3(H  / 32, H / 32), dim3(32, 8), 0, stream>>>(w0,   w0t,   H,  H);
    k_cvt_t<<<dim3(H  / 32, H / 32), dim3(32, 8), 0, stream>>>(w1,   w1t,   H,  H);
    k_cvt_t<<<dim3(H  / 32, H / 32), dim3(32, 8), 0, stream>>>(cw,   cwt,   H,  H);
    k_cvt_t<<<dim3(DO / 32, H / 32), dim3(32, 8), 0, stream>>>(wout, woutt, DO, H);

    // transcendent probs
    k_probs<<<H, 256, 0, stream>>>(tw, probs);

    // layer 0: h1 = relu(x @ w0 + b0)
    gemm_bf16<0><<<dim3(Bq / BM, H / BN), 256, 0, stream>>>(
        xb, w0t, b0, nullptr, h1b, Bq, H, H);
    // layer 1: h2 = relu(h1 @ w1 + b1)
    gemm_bf16<0><<<dim3(Bq / BM, H / BN), 256, 0, stream>>>(
        h1b, w1t, b1, nullptr, h2b, Bq, H, H);
    // layer 2: h3 = tanh(h2 @ cw + cb + probs)
    gemm_bf16<2><<<dim3(Bq / BM, H / BN), 256, 0, stream>>>(
        h2b, cwt, cb, probs, h3b, Bq, H, H);
    // layer 3: out = relu(h3 @ wout + bout)  (f32)
    gemm_bf16<3><<<dim3(Bq / BM, DO / BN), 256, 0, stream>>>(
        h3b, woutt, bout, nullptr, d_out, Bq, DO, H);
}
